// BinaryClassifyModel__29446295781402
// MI455X (gfx1250) — matrix operand order repaired and checked by reading, not compiled, not run on hardware
//
#include <hip/hip_runtime.h>
#include <hip/hip_bf16.h>
#include <math.h>
#include <stdint.h>

// ---------------------------------------------------------------------------
// 6-layer causal transformer forward for MI455X (gfx1250, wave32).
// GEMMs: v_wmma_f32_16x16x32_bf16, 128x128 block tile, 32x64 per wave,
// double-buffered LDS staging via GLOBAL_LOAD_ASYNC_TO_LDS_B128 when the
// toolchain exposes it (ASYNCcnt + s_wait_asynccnt), plain loads otherwise.
// All B operands are [N,K] row-major so async copies need no transpose.
// ---------------------------------------------------------------------------

constexpr int kB = 8, kS = 2048, kH = 768, kFF = 3072, kL = 6;

typedef __attribute__((ext_vector_type(16))) __bf16 v16bf;
typedef __attribute__((ext_vector_type(8)))  float  v8f;
typedef unsigned short u16;
typedef unsigned int   u32;

#if defined(__AMDGCN__) && __has_builtin(__builtin_amdgcn_global_load_async_to_lds_b128)
#define TFX_ASYNC 1
#else
#define TFX_ASYNC 0
#endif

__device__ __forceinline__ u16 f2bf(float f) {
  u32 u = __float_as_uint(f);
  u += 0x7fffu + ((u >> 16) & 1u);   // round-to-nearest-even
  return (u16)(u >> 16);
}

#if TFX_ASYNC
typedef int tfx_v4i __attribute__((vector_size(4 * sizeof(int))));
__device__ __forceinline__ void tfx_cp16(const u16* g, u16* l) {
  __builtin_amdgcn_global_load_async_to_lds_b128(
      (__attribute__((address_space(1))) tfx_v4i*)g,
      (__attribute__((address_space(3))) tfx_v4i*)l, 0, 0);
}
#endif

__device__ __forceinline__ void tfx_wait_async() {
#if TFX_ASYNC
#if __has_builtin(__builtin_amdgcn_s_wait_asynccnt)
  __builtin_amdgcn_s_wait_asynccnt(0);
#else
  asm volatile("s_wait_asynccnt 0x0" ::: "memory");
#endif
#endif
}

// ---------------------------------------------------------------------------
// f32 -> bf16 transposed per slice (weights)
// ---------------------------------------------------------------------------
__global__ __launch_bounds__(256) void tfx_cvt_bf16_T(const float* __restrict__ in,
                                                      u16* __restrict__ out,
                                                      int R, int C, long long total) {
  long long i = (long long)blockIdx.x * 256 + threadIdx.x;
  if (i >= total) return;
  long long rc = (long long)R * C;
  int l = (int)(i / rc);
  int rem = (int)(i - (long long)l * rc);
  int r = rem / C, c = rem - r * C;
  out[(long long)l * rc + (long long)c * R + r] = f2bf(in[i]);
}

// ---------------------------------------------------------------------------
// x = tok_emb[ids] + pos_emb
// ---------------------------------------------------------------------------
__global__ __launch_bounds__(256) void tfx_embed(const long long* __restrict__ ids,
                                                 const float* __restrict__ tok,
                                                 const float* __restrict__ pos,
                                                 float* __restrict__ x) {
  long long i = (long long)blockIdx.x * 256 + threadIdx.x;
  const long long total = (long long)kB * kS * kH;
  if (i >= total) return;
  int h = (int)(i % kH);
  long long bs = i / kH;
  int s = (int)(bs % kS);
  long long v = ids[bs];
  x[i] = tok[v * kH + h] + pos[(long long)s * kH + h];
}

// ---------------------------------------------------------------------------
// Row LayerNorm (f32 in) -> bf16 out.  One wave32 per row.
// ---------------------------------------------------------------------------
__global__ __launch_bounds__(256) void tfx_layernorm_bf16(const float* __restrict__ x,
                                                          const float* __restrict__ g,
                                                          const float* __restrict__ b,
                                                          u16* __restrict__ out,
                                                          int nrows) {
  int lane = threadIdx.x & 31;
  int row  = blockIdx.x * 8 + (threadIdx.x >> 5);
  if (row >= nrows) return;
  const float* xr = x + (size_t)row * kH;
  float s = 0.f, ss = 0.f;
  for (int h = lane; h < kH; h += 32) { float v = xr[h]; s += v; ss += v * v; }
  for (int o = 16; o; o >>= 1) { s += __shfl_xor(s, o); ss += __shfl_xor(ss, o); }
  float mean = s * (1.0f / kH);
  float var  = ss * (1.0f / kH) - mean * mean;
  float rstd = rsqrtf(var + 1e-5f);
  u16* orow = out + (size_t)row * kH;
  for (int h = lane; h < kH; h += 32)
    orow[h] = f2bf((xr[h] - mean) * rstd * g[h] + b[h]);
}

// ---------------------------------------------------------------------------
// bf16 WMMA GEMM:  C[M,N] = A[M,K] @ B^T  with B stored [N,K] row-major.
// Block tile 128x128, BK=32, 8 waves (4x2), 32x64 per wave = 2x4 WMMA tiles.
// Double-buffered LDS staging (async global->LDS when available).
// Epilogue: +bias, exact GELU, +residual; stores f32 / bf16 / bf16-transposed.
// ---------------------------------------------------------------------------
__global__ __launch_bounds__(256) void tfx_gemm_wmma(
    const u16* __restrict__ A, const u16* __restrict__ Bm,
    const float* __restrict__ bias, const float* __restrict__ resid,
    float* __restrict__ outF, u16* __restrict__ outB, u16* __restrict__ outBT,
    int M, int N, int K, int ldb, int act_gelu) {
  constexpr int LDA = 48;                 // padded row stride (ushorts), 96B
  constexpr int TILE = 128 * LDA;         // one 128x32 tile (ushorts)
  __shared__ alignas(16) u16 smem[4 * TILE];   // A0,A1,B0,B1 = 48 KiB
  const int t    = threadIdx.x;
  const int lane = t & 31, wave = t >> 5;
  const int bm = blockIdx.y * 128, bn = blockIdx.x * 128;
  const int wm = (wave >> 1) * 32, wn = (wave & 1) * 64;

  // stage one 128x32 A tile + 128x32 B tile into buffer `buf`
  auto stage = [&](int k0, int buf) {
    u16* Ab = smem + buf * TILE;
    u16* Bb = smem + 2 * TILE + buf * TILE;
    const int r0 = t >> 2, c0 = t & 3;          // chunks t and t+256 of 512
    const int r1 = (t + 256) >> 2, c1 = t & 3;
    const u16* gA0 = A  + (size_t)(bm + r0) * K   + k0 + c0 * 8;
    const u16* gA1 = A  + (size_t)(bm + r1) * K   + k0 + c1 * 8;
    const u16* gB0 = Bm + (size_t)(bn + r0) * ldb + k0 + c0 * 8;
    const u16* gB1 = Bm + (size_t)(bn + r1) * ldb + k0 + c1 * 8;
    u16* lA0 = Ab + r0 * LDA + c0 * 8;
    u16* lA1 = Ab + r1 * LDA + c1 * 8;
    u16* lB0 = Bb + r0 * LDA + c0 * 8;
    u16* lB1 = Bb + r1 * LDA + c1 * 8;
#if TFX_ASYNC
    tfx_cp16(gA0, lA0);
    tfx_cp16(gA1, lA1);
    tfx_cp16(gB0, lB0);
    tfx_cp16(gB1, lB1);
#else
    *reinterpret_cast<uint4*>(lA0) = *reinterpret_cast<const uint4*>(gA0);
    *reinterpret_cast<uint4*>(lA1) = *reinterpret_cast<const uint4*>(gA1);
    *reinterpret_cast<uint4*>(lB0) = *reinterpret_cast<const uint4*>(gB0);
    *reinterpret_cast<uint4*>(lB1) = *reinterpret_cast<const uint4*>(gB1);
#endif
  };

  v8f acc[2][4];
#pragma unroll
  for (int i = 0; i < 2; i++)
#pragma unroll
    for (int j = 0; j < 4; j++)
#pragma unroll
      for (int e = 0; e < 8; e++) acc[i][j][e] = 0.0f;

  const int nk = K >> 5;
  stage(0, 0);
  tfx_wait_async();
  __syncthreads();

  for (int ks = 0; ks < nk; ks++) {
    const int cur = ks & 1;
    if (ks + 1 < nk) stage((ks + 1) << 5, cur ^ 1);

    const u16* Ab = smem + cur * TILE;
    const u16* Bb = smem + 2 * TILE + cur * TILE;
    union Frag { v16bf v; u32 u[8]; };
    Frag af[2], bf[4];
    const int kb_a = (lane >> 4) * 8;      // ISA 7.12.2 16-bit A layout
    const int kb_b = (lane >> 4) * 8;      // B layout matches A (table order)
#pragma unroll
    for (int i = 0; i < 2; i++) {
      const u16* ap = Ab + (wm + i * 16 + (lane & 15)) * LDA;
#pragma unroll
      for (int v = 0; v < 8; v++) {
        int k = (v < 4) ? (kb_a + 2 * v) : (16 + kb_a + 2 * (v - 4));
        af[i].u[v] = *reinterpret_cast<const u32*>(ap + k);
      }
    }
#pragma unroll
    for (int j = 0; j < 4; j++) {
      const u16* bp = Bb + (wn + j * 16 + (lane & 15)) * LDA;
#pragma unroll
      for (int v = 0; v < 8; v++) {
        int k = (v < 4) ? (kb_b + 2 * v) : (16 + kb_b + 2 * (v - 4));
        bf[j].u[v] = *reinterpret_cast<const u32*>(bp + k);
      }
    }
#pragma unroll
    for (int i = 0; i < 2; i++)
#pragma unroll
      for (int j = 0; j < 4; j++)
        acc[i][j] = __builtin_amdgcn_wmma_f32_16x16x32_bf16(
            false, af[i].v, false, bf[j].v, (short)0, acc[i][j], false, false);

    tfx_wait_async();
    __syncthreads();
  }

  // epilogue: C layout = VGPR r -> M=r (lanes 0-15) / M=8+r (lanes 16-31)
#pragma unroll
  for (int i = 0; i < 2; i++) {
    int gm0 = bm + wm + i * 16 + (lane >> 4) * 8;
#pragma unroll
    for (int j = 0; j < 4; j++) {
      int gn = bn + wn + j * 16 + (lane & 15);
      float bv = bias ? bias[gn] : 0.0f;
#pragma unroll
      for (int r = 0; r < 8; r++) {
        int gm = gm0 + r;
        float val = acc[i][j][r] + bv;
        if (act_gelu) val = 0.5f * val * (1.0f + erff(val * 0.70710678118654752f));
        size_t o = (size_t)gm * N + gn;
        if (resid) val += resid[o];
        if (outF)  outF[o] = val;
        if (outB)  outB[o] = f2bf(val);
        if (outBT) outBT[(size_t)gn * M + gm] = f2bf(val);
      }
    }
  }
}

// ---------------------------------------------------------------------------
// Masked causal softmax over one row of scores -> bf16 probabilities
// ---------------------------------------------------------------------------
__global__ __launch_bounds__(256) void tfx_softmax(const float* __restrict__ sc,
                                                   u16* __restrict__ pr,
                                                   const int* __restrict__ amask,
                                                   float scale) {
  const int q = blockIdx.x;
  const float* srow = sc + (size_t)q * kS;
  u16* prow = pr + (size_t)q * kS;
  __shared__ float red[8];
  int lane = threadIdx.x & 31, wave = threadIdx.x >> 5;
  float m = -3.0e38f;
  for (int c = threadIdx.x; c <= q; c += 256)
    if (amask[c] != 0) m = fmaxf(m, srow[c] * scale);
  for (int o = 16; o; o >>= 1) m = fmaxf(m, __shfl_xor(m, o));
  if (lane == 0) red[wave] = m;
  __syncthreads();
  m = red[0];
  for (int i = 1; i < 8; i++) m = fmaxf(m, red[i]);
  float s = 0.f;
  for (int c = threadIdx.x; c <= q; c += 256)
    if (amask[c] != 0) s += expf(srow[c] * scale - m);
  for (int o = 16; o; o >>= 1) s += __shfl_xor(s, o);
  __syncthreads();
  if (lane == 0) red[wave] = s;
  __syncthreads();
  s = 0.f;
  for (int i = 0; i < 8; i++) s += red[i];
  float inv = (s > 0.f) ? 1.0f / s : 0.0f;
  for (int c = threadIdx.x; c < kS; c += 256) {
    float p = 0.f;
    if (c <= q && amask[c] != 0) p = expf(srow[c] * scale - m) * inv;
    prow[c] = f2bf(p);
  }
}

// ---------------------------------------------------------------------------
// Final LN on the last non-padded token + classifier dot.  grid = B blocks.
// ---------------------------------------------------------------------------
__device__ __forceinline__ float tfx_bsum(float v, float* red) {
  int lane = threadIdx.x & 31, wave = threadIdx.x >> 5;
  for (int o = 16; o; o >>= 1) v += __shfl_xor(v, o);
  __syncthreads();
  if (lane == 0) red[wave] = v;
  __syncthreads();
  float r = 0.f;
  for (int i = 0; i < 8; i++) r += red[i];
  return r;
}

__global__ __launch_bounds__(256) void tfx_final_head(const float* __restrict__ x,
                                                      const int* __restrict__ amask,
                                                      const float* __restrict__ g,
                                                      const float* __restrict__ b,
                                                      const float* __restrict__ clsW,
                                                      const float* __restrict__ clsb,
                                                      float* __restrict__ out) {
  const int bb = blockIdx.x;
  __shared__ float red[8];
  float cnt = 0.f;
  for (int c = threadIdx.x; c < kS; c += 256)
    cnt += (amask[bb * kS + c] != 0) ? 1.0f : 0.0f;
  cnt = tfx_bsum(cnt, red);
  int last = (int)cnt - 1;
  const float* xr = x + ((size_t)bb * kS + last) * kH;
  float s = 0.f, ss = 0.f;
  for (int h = threadIdx.x; h < kH; h += 256) { float v = xr[h]; s += v; ss += v * v; }
  s = tfx_bsum(s, red);
  __syncthreads();
  ss = tfx_bsum(ss, red);
  float mean = s * (1.0f / kH);
  float var  = ss * (1.0f / kH) - mean * mean;
  float rstd = rsqrtf(var + 1e-5f);
  float d = 0.f;
  for (int h = threadIdx.x; h < kH; h += 256)
    d += ((xr[h] - mean) * rstd * g[h] + b[h]) * clsW[h];
  __syncthreads();
  d = tfx_bsum(d, red);
  if (threadIdx.x == 0) out[bb] = d + clsb[0];
}

// ---------------------------------------------------------------------------
// Host orchestration
// ---------------------------------------------------------------------------
static inline void launch_gemm(hipStream_t st,
                               const u16* A, const u16* Bm,
                               const float* bias, const float* resid,
                               float* outF, u16* outB, u16* outBT,
                               int M, int N, int K, int ldb, int gelu) {
  dim3 g(N / 128, M / 128);
  tfx_gemm_wmma<<<g, 256, 0, st>>>(A, Bm, bias, resid, outF, outB, outBT,
                                   M, N, K, ldb, gelu);
}

extern "C" void kernel_launch(void* const* d_in, const int* in_sizes, int n_in,
                              void* d_out, int out_size, void* d_ws, size_t ws_size,
                              hipStream_t stream) {
  (void)in_sizes; (void)n_in; (void)out_size; (void)ws_size;
  const long long* ids   = (const long long*)d_in[0];
  const int*   amask     = (const int*)  d_in[1];
  const float* tok_emb   = (const float*)d_in[2];
  const float* pos_emb   = (const float*)d_in[3];
  const float* Wq = (const float*)d_in[4],  *bq = (const float*)d_in[5];
  const float* Wk = (const float*)d_in[6],  *bk = (const float*)d_in[7];
  const float* Wv = (const float*)d_in[8],  *bv = (const float*)d_in[9];
  const float* Wo = (const float*)d_in[10], *bo = (const float*)d_in[11];
  const float* ln1_g = (const float*)d_in[12], *ln1_b = (const float*)d_in[13];
  const float* ln2_g = (const float*)d_in[14], *ln2_b = (const float*)d_in[15];
  const float* W1 = (const float*)d_in[16], *b1 = (const float*)d_in[17];
  const float* W2 = (const float*)d_in[18], *b2 = (const float*)d_in[19];
  const float* fln_g = (const float*)d_in[20], *fln_b = (const float*)d_in[21];
  const float* clsW  = (const float*)d_in[22], *clsb  = (const float*)d_in[23];

  char* ws = (char*)d_ws;
  size_t off = 0;
  auto alloc = [&](size_t bytes) -> void* {
    void* p = ws + off;
    off = (off + bytes + 255) & ~(size_t)255;
    return p;
  };
  const size_t BSH = (size_t)kB * kS * kH;
  float* x   = (float*)alloc(BSH * 4);
  u16*   hbf = (u16*)  alloc(BSH * 2);
  u16*   qb  = (u16*)  alloc(BSH * 2);
  u16*   kbf = (u16*)  alloc(BSH * 2);      // K: natural [BS,H] == [N,K] per batch
  u16*   vT  = (u16*)  alloc(BSH * 2);      // V^T: [H, BS]
  u16*   ao  = (u16*)  alloc(BSH * 2);
  float* sc  = (float*)alloc((size_t)kS * kS * 4);
  u16*   pr  = (u16*)  alloc((size_t)kS * kS * 2);
  u16*   ffn = (u16*)  alloc((size_t)kB * kS * kFF * 2);
  u16* wqT = (u16*)alloc((size_t)kL * kH * kH * 2);   // all weights bf16, [N,K]
  u16* wkT = (u16*)alloc((size_t)kL * kH * kH * 2);
  u16* wvT = (u16*)alloc((size_t)kL * kH * kH * 2);
  u16* woT = (u16*)alloc((size_t)kL * kH * kH * 2);
  u16* w1T = (u16*)alloc((size_t)kL * kH * kFF * 2);
  u16* w2T = (u16*)alloc((size_t)kL * kFF * kH * 2);

  auto cvtT = [&](const float* src, u16* dst, int R, int C) {
    long long n = (long long)kL * R * C;
    tfx_cvt_bf16_T<<<(unsigned)((n + 255) / 256), 256, 0, stream>>>(src, dst, R, C, n);
  };
  cvtT(Wq, wqT, kH, kH);
  cvtT(Wk, wkT, kH, kH);
  cvtT(Wv, wvT, kH, kH);
  cvtT(Wo, woT, kH, kH);
  cvtT(W1, w1T, kH, kFF);
  cvtT(W2, w2T, kFF, kH);

  const long long tot = (long long)kB * kS * kH;
  tfx_embed<<<(unsigned)((tot + 255) / 256), 256, 0, stream>>>(ids, tok_emb, pos_emb, x);

  const int BS = kB * kS;
  const float scale = 1.0f / sqrtf((float)kH);

  for (int l = 0; l < kL; l++) {
    const size_t wHH = (size_t)l * kH * kH;
    // LN1 -> hbf (bf16)
    tfx_layernorm_bf16<<<BS / 8, 256, 0, stream>>>(x, ln1_g + l * kH, ln1_b + l * kH, hbf, BS);
    // Q, K, V projections (V stored transposed for probs@V)
    launch_gemm(stream, hbf, wqT + wHH, bq + l * kH, nullptr, nullptr, qb,  nullptr, BS, kH, kH, kH, 0);
    launch_gemm(stream, hbf, wkT + wHH, bk + l * kH, nullptr, nullptr, kbf, nullptr, BS, kH, kH, kH, 0);
    launch_gemm(stream, hbf, wvT + wHH, bv + l * kH, nullptr, nullptr, nullptr, vT,  BS, kH, kH, kH, 0);
    // attention per batch (reuse S x S scratch)
    for (int b = 0; b < kB; b++) {
      const size_t bo_ = (size_t)b * kS * kH;
      launch_gemm(stream, qb + bo_, kbf + bo_, nullptr, nullptr, sc, nullptr, nullptr,
                  kS, kS, kH, kH, 0);
      tfx_softmax<<<kS, 256, 0, stream>>>(sc, pr, amask + b * kS, scale);
      launch_gemm(stream, pr, vT + (size_t)b * kS, nullptr, nullptr, nullptr, ao + bo_, nullptr,
                  kS, kH, kS, BS, 0);
    }
    // O projection + residual into x
    launch_gemm(stream, ao, woT + wHH, bo + l * kH, x, x, nullptr, nullptr, BS, kH, kH, kH, 0);
    // LN2 -> hbf
    tfx_layernorm_bf16<<<BS / 8, 256, 0, stream>>>(x, ln2_g + l * kH, ln2_b + l * kH, hbf, BS);
    // FFN
    launch_gemm(stream, hbf, w1T + (size_t)l * kH * kFF, b1 + l * kFF, nullptr, nullptr, ffn, nullptr,
                BS, kFF, kH, kH, 1);
    launch_gemm(stream, ffn, w2T + (size_t)l * kFF * kH, b2 + l * kH, x, x, nullptr, nullptr,
                BS, kH, kFF, kFF, 0);
  }

  tfx_final_head<<<kB, 256, 0, stream>>>(x, amask, fln_g, fln_b, clsW, clsb, (float*)d_out);
}
